// Encoder_37924561223950
// MI455X (gfx1250) — matrix operand order repaired and checked by reading, not compiled, not run on hardware
//
#include <hip/hip_runtime.h>

#define NL   12
#define BS   8
#define TT   512
#define DD   512
#define HH   8
#define DKK  64
#define DFFN 2048
#define KC   15
#define NTOK (BS*TT)
#define LNEPS 1e-5f

typedef __bf16 bf16;
typedef __attribute__((ext_vector_type(4)))  __bf16 v4bf;
typedef __attribute__((ext_vector_type(8)))  __bf16 v8bf;
typedef __attribute__((ext_vector_type(16))) __bf16 v16bf;
typedef __attribute__((ext_vector_type(8)))  float  v8f;

// ---------------------------------------------------------------------------
// WMMA GEMM:  C = [resid +] resScale * act(A @ B [+ bias])
//   A: (M,K) row-major, row stride lda
//   B: TRANSB==0 -> (K,N) row-major ; TRANSB==1 -> (N,K) row-major (A @ B^T)
//   batched over grid.z = nb*nh with separate (b,h) offsets
// Block tile 128x64x32, 128 threads (4 waves stacked in M).
// Each wave: 32 rows x 64 cols = 2x4 WMMA fragments = 8 WMMAs per K-step.
// Double-buffered LDS; next K-tile staged in registers across the WMMA issue.
// ---------------------------------------------------------------------------
template<int TRANSB, int RELU, int HASB, int HASR, int NT>
__launch_bounds__(128)
__global__ void gemm_wmma_kernel(
    const float* __restrict__ A, const float* __restrict__ Bm,
    const float* __restrict__ bias, const float* resid, float* C,
    int M, int N, int Kd, int lda, int ldb, int ldc,
    long sAb, long sAh, long sBb, long sBh, long sCb, long sCh,
    int Hb, float resScale)
{
    __shared__ __align__(16) bf16 As[2][128][40];  // pitch 40 -> 80B rows, 16B aligned
    __shared__ __align__(16) bf16 Bs[2][64][40];   // stored N-major in LDS

    const int z  = blockIdx.z;
    const int bb = z / Hb, hh = z % Hb;
    A  += (size_t)bb * sAb + (size_t)hh * sAh;
    Bm += (size_t)bb * sBb + (size_t)hh * sBh;
    const size_t coff = (size_t)bb * sCb + (size_t)hh * sCh;
    C += coff;
    if (HASR) resid += coff;

    const int m0   = blockIdx.y * 128;
    const int n0   = blockIdx.x * 64;
    const int tid  = threadIdx.x;
    const int lane = tid & 31;
    const int wave = tid >> 5;
    const int wm   = wave * 32;          // wave row offset within 128-row tile
    const int hf   = lane >> 4;
    const int lm   = lane & 15;

    // staging maps
    const int ar  = tid >> 1, ak = (tid & 1) * 16;   // A/TRANSB-B: row, k-chunk
    const int bn4 = (tid & 15) * 4;                  // NN B: 4x4 (k x n) block
    const int bk4 = (tid >> 4) * 4;

    float4 fa[8], fb[4];

    auto load_tile = [&](int k0) {
        const float* srcA0 = A + (size_t)(m0 + ar) * lda + (k0 + ak);
        const float* srcA1 = A + (size_t)(m0 + ar + 64) * lda + (k0 + ak);
        #pragma unroll
        for (int j = 0; j < 4; j++) fa[j]     = *(const float4*)(srcA0 + 4 * j);
        #pragma unroll
        for (int j = 0; j < 4; j++) fa[4 + j] = *(const float4*)(srcA1 + 4 * j);
        if (k0 + 32 < Kd) {
            __builtin_prefetch((const void*)(srcA0 + 32), 0, 3);
            __builtin_prefetch((const void*)(srcA1 + 32), 0, 3);
        }
        if (TRANSB) {
            const float* srcB = Bm + (size_t)(n0 + ar) * ldb + (k0 + ak);
            #pragma unroll
            for (int j = 0; j < 4; j++) fb[j] = *(const float4*)(srcB + 4 * j);
            if (k0 + 32 < Kd) __builtin_prefetch((const void*)(srcB + 32), 0, 3);
        } else {
            #pragma unroll
            for (int j = 0; j < 4; j++)
                fb[j] = *(const float4*)(Bm + (size_t)(k0 + bk4 + j) * ldb + (n0 + bn4));
            if (k0 + 32 < Kd)
                __builtin_prefetch((const void*)(Bm + (size_t)(k0 + 32 + bk4) * ldb + (n0 + bn4)), 0, 3);
        }
    };

    auto store_tile = [&](int buf) {
        #pragma unroll
        for (int j = 0; j < 4; j++) {
            As[buf][ar][ak + 4 * j + 0] = (bf16)fa[j].x;
            As[buf][ar][ak + 4 * j + 1] = (bf16)fa[j].y;
            As[buf][ar][ak + 4 * j + 2] = (bf16)fa[j].z;
            As[buf][ar][ak + 4 * j + 3] = (bf16)fa[j].w;
        }
        #pragma unroll
        for (int j = 0; j < 4; j++) {
            As[buf][ar + 64][ak + 4 * j + 0] = (bf16)fa[4 + j].x;
            As[buf][ar + 64][ak + 4 * j + 1] = (bf16)fa[4 + j].y;
            As[buf][ar + 64][ak + 4 * j + 2] = (bf16)fa[4 + j].z;
            As[buf][ar + 64][ak + 4 * j + 3] = (bf16)fa[4 + j].w;
        }
        if (TRANSB) {
            #pragma unroll
            for (int j = 0; j < 4; j++) {
                Bs[buf][ar][ak + 4 * j + 0] = (bf16)fb[j].x;
                Bs[buf][ar][ak + 4 * j + 1] = (bf16)fb[j].y;
                Bs[buf][ar][ak + 4 * j + 2] = (bf16)fb[j].z;
                Bs[buf][ar][ak + 4 * j + 3] = (bf16)fb[j].w;
            }
        } else {
            // in-register 4x4 transpose -> 4 contiguous 8B stores (ds_store_b64)
            v4bf p0 = { (bf16)fb[0].x, (bf16)fb[1].x, (bf16)fb[2].x, (bf16)fb[3].x };
            v4bf p1 = { (bf16)fb[0].y, (bf16)fb[1].y, (bf16)fb[2].y, (bf16)fb[3].y };
            v4bf p2 = { (bf16)fb[0].z, (bf16)fb[1].z, (bf16)fb[2].z, (bf16)fb[3].z };
            v4bf p3 = { (bf16)fb[0].w, (bf16)fb[1].w, (bf16)fb[2].w, (bf16)fb[3].w };
            *(v4bf*)&Bs[buf][bn4 + 0][bk4] = p0;
            *(v4bf*)&Bs[buf][bn4 + 1][bk4] = p1;
            *(v4bf*)&Bs[buf][bn4 + 2][bk4] = p2;
            *(v4bf*)&Bs[buf][bn4 + 3][bk4] = p3;
        }
    };

    v8f acc[2][4] = {};

    load_tile(0);
    store_tile(0);
    __syncthreads();

    const int nk = Kd >> 5;
    for (int it = 0; it < nk; it++) {
        const int cur = it & 1;
        const bool more = (it + 1) < nk;
        if (more) load_tile((it + 1) << 5);   // global loads in flight across WMMAs

        // fragments per ISA 7.12.2 lane layout
        v16bf afrag[2], bfrag[4];
        #pragma unroll
        for (int ti = 0; ti < 2; ti++) {
            const bf16* pr = &As[cur][wm + ti * 16 + lm][0];
            v8bf a0 = *(const v8bf*)(pr + hf * 8);       // K 0-7 / 8-15
            v8bf a1 = *(const v8bf*)(pr + 16 + hf * 8);  // K 16-23 / 24-31
            afrag[ti] = __builtin_shufflevector(a0, a1,
                0,1,2,3,4,5,6,7,8,9,10,11,12,13,14,15);
        }
        #pragma unroll
        for (int tj = 0; tj < 4; tj++) {
            const bf16* pr = &Bs[cur][tj * 16 + lm][0];
            v8bf b0 = *(const v8bf*)(pr + hf * 8);
            v8bf b1 = *(const v8bf*)(pr + 16 + hf * 8);
            bfrag[tj] = __builtin_shufflevector(b0, b1,
                0,1,2,3,4,5,6,7,8,9,10,11,12,13,14,15);
        }
        #pragma unroll
        for (int ti = 0; ti < 2; ti++)
            #pragma unroll
            for (int tj = 0; tj < 4; tj++)
                acc[ti][tj] = __builtin_amdgcn_wmma_f32_16x16x32_bf16(
                    false, afrag[ti], false, bfrag[tj],
                    (short)0, acc[ti][tj], false, false);

        if (more) store_tile(cur ^ 1);
        __syncthreads();
    }

    // epilogue: straight-line (bias/resid compiled in or out)
    #pragma unroll
    for (int ti = 0; ti < 2; ti++) {
        #pragma unroll
        for (int tj = 0; tj < 4; tj++) {
            const int col = n0 + tj * 16 + lm;
            const float bv = HASB ? bias[col] : 0.0f;
            #pragma unroll
            for (int vg = 0; vg < 8; vg++) {
                const int row = m0 + wm + ti * 16 + hf * 8 + vg;
                float v = acc[ti][tj][vg] + bv;
                if (RELU) v = fmaxf(v, 0.0f);
                v *= resScale;
                const size_t idx = (size_t)row * ldc + col;
                if (HASR) v += resid[idx];
                if (NT) __builtin_nontemporal_store(v, &C[idx]);
                else    C[idx] = v;
            }
        }
    }
}

static inline void launch_gemm(hipStream_t st,
    const float* A, const float* B, const float* bias, const float* resid,
    float* C, int M, int N, int K, int lda, int ldb, int ldc,
    long sAb, long sAh, long sBb, long sBh, long sCb, long sCh,
    int nb, int nh, int transb, int relu, float resScale, int nt)
{
    dim3 g(N / 64, M / 128, nb * nh), blk(128);
    const int hb = bias ? 1 : 0, hr = resid ? 1 : 0;
    if (nt && transb && !relu && !hb && !hr) {   // score GEMMs: NT stores
        gemm_wmma_kernel<1,0,0,0,1><<<g, blk, 0, st>>>(A, B, bias, resid, C,
            M, N, K, lda, ldb, ldc, sAb, sAh, sBb, sBh, sCb, sCh, nh, resScale);
        return;
    }
    const int sel = (transb << 3) | (relu << 2) | (hb << 1) | hr;
#define GEMM_CASE(TB, RL, HB, HR)                                              \
    case ((TB << 3) | (RL << 2) | (HB << 1) | HR):                             \
        gemm_wmma_kernel<TB, RL, HB, HR, 0><<<g, blk, 0, st>>>(A, B, bias,     \
            resid, C, M, N, K, lda, ldb, ldc, sAb, sAh, sBb, sBh, sCb, sCh,    \
            nh, resScale);                                                     \
        break;
    switch (sel) {
        GEMM_CASE(0,0,0,0) GEMM_CASE(0,0,0,1) GEMM_CASE(0,0,1,0) GEMM_CASE(0,0,1,1)
        GEMM_CASE(0,1,0,0) GEMM_CASE(0,1,0,1) GEMM_CASE(0,1,1,0) GEMM_CASE(0,1,1,1)
        GEMM_CASE(1,0,0,0) GEMM_CASE(1,0,0,1) GEMM_CASE(1,0,1,0) GEMM_CASE(1,0,1,1)
        GEMM_CASE(1,1,0,0) GEMM_CASE(1,1,0,1) GEMM_CASE(1,1,1,0) GEMM_CASE(1,1,1,1)
    }
#undef GEMM_CASE
}

// ---------------------------------------------------------------------------
// LayerNorm: one wave32 per row of 512
// ---------------------------------------------------------------------------
__launch_bounds__(256)
__global__ void ln_kernel(const float* __restrict__ x,
                          const float* __restrict__ g,
                          const float* __restrict__ b,
                          float* __restrict__ y)
{
    const int row  = blockIdx.x * 8 + (threadIdx.x >> 5);
    const int lane = threadIdx.x & 31;
    const float* xr = x + (size_t)row * DD;
    float s = 0.f, ss = 0.f;
    for (int c = lane; c < DD; c += 32) { float v = xr[c]; s += v; ss += v * v; }
    #pragma unroll
    for (int o = 16; o > 0; o >>= 1) {
        s  += __shfl_xor(s,  o, 32);
        ss += __shfl_xor(ss, o, 32);
    }
    const float m   = s  * (1.0f / DD);
    const float var = ss * (1.0f / DD) - m * m;
    const float inv = rsqrtf(var + LNEPS);
    float* yr = y + (size_t)row * DD;
    for (int c = lane; c < DD; c += 32)
        yr[c] = (xr[c] - m) * inv * g[c] + b[c];
}

// sinusoidal positional table pe[t, c]
__global__ void pe_kernel(float* __restrict__ pe)
{
    const int i = blockIdx.x * blockDim.x + threadIdx.x;
    if (i >= TT * DD) return;
    const int t = i >> 9, c = i & 511;
    const float f   = __expf((float)(c & ~1) * (-9.21034037198f / 512.0f));
    const float ang = (float)t * f;
    pe[i] = (c & 1) ? __cosf(ang) : __sinf(ang);
}

// qu = q + pbu ; qv = q + pbv  (broadcast over rows, length 512)
__global__ void addpb_kernel(const float* __restrict__ q,
                             const float* __restrict__ pbu,
                             const float* __restrict__ pbv,
                             float* __restrict__ qu, float* __restrict__ qv)
{
    const int i = blockIdx.x * blockDim.x + threadIdx.x;
    const int c = i & 511;
    const float v = q[i];
    qu[i] = v + pbu[c];
    qv[i] = v + pbv[c];
}

// softmax over keys with legacy Transformer-XL relative shift of bd
__launch_bounds__(256)
__global__ void attn_softmax_kernel(const float* __restrict__ ac,
                                    const float* __restrict__ bd,
                                    float* __restrict__ out, float scale)
{
    const int q = blockIdx.x;
    const int z = blockIdx.y;
    const int tid = threadIdx.x;
    const float* acr = ac + ((size_t)z * TT + q) * TT;
    const float* bdz = bd + (size_t)z * TT * TT;
    float* o = out + ((size_t)z * TT + q) * TT;

    __shared__ float sbuf[TT];
    __shared__ float red[8];

    float lmax = -3.4e38f;
    for (int s = tid; s < TT; s += 256) {
        float bdv;
        if (s <= q)          bdv = __builtin_nontemporal_load(&bdz[(size_t)q * TT + (TT - q + s - 1)]);
        else if (s == q + 1) bdv = 0.0f;
        else                 bdv = __builtin_nontemporal_load(&bdz[(size_t)(q + 1) * TT + (s - q - 2)]);
        const float acv = __builtin_nontemporal_load(&acr[s]);
        const float val = (acv + bdv) * scale;
        sbuf[s] = val;
        lmax = fmaxf(lmax, val);
    }
    #pragma unroll
    for (int off = 16; off > 0; off >>= 1) lmax = fmaxf(lmax, __shfl_xor(lmax, off, 32));
    if ((tid & 31) == 0) red[tid >> 5] = lmax;
    __syncthreads();
    float bmax = red[0];
    #pragma unroll
    for (int w = 1; w < 8; w++) bmax = fmaxf(bmax, red[w]);
    __syncthreads();

    float lsum = 0.f;
    for (int s = tid; s < TT; s += 256) {
        const float e = __expf(sbuf[s] - bmax);
        sbuf[s] = e;
        lsum += e;
    }
    #pragma unroll
    for (int off = 16; off > 0; off >>= 1) lsum += __shfl_xor(lsum, off, 32);
    if ((tid & 31) == 0) red[tid >> 5] = lsum;
    __syncthreads();
    float bsum = 0.f;
    #pragma unroll
    for (int w = 0; w < 8; w++) bsum += red[w];
    const float inv = 1.0f / bsum;
    for (int s = tid; s < TT; s += 256) o[s] = sbuf[s] * inv;
}

// GLU: h (NTOK, 1024) -> out (NTOK, 512)
__global__ void glu_kernel(const float* __restrict__ h, float* __restrict__ out)
{
    const int i = blockIdx.x * blockDim.x + threadIdx.x;
    const int row = i >> 9, c = i & 511;
    const float a = h[(size_t)row * 1024 + c];
    const float g = h[(size_t)row * 1024 + 512 + c];
    out[i] = a * (1.0f / (1.0f + __expf(-g)));
}

// depthwise conv over time, K=15, same padding; layout (B,T,C)
__global__ void dwconv_kernel(const float* __restrict__ x,
                              const float* __restrict__ w,
                              const float* __restrict__ bias,
                              float* __restrict__ y)
{
    const int i = blockIdx.x * blockDim.x + threadIdx.x;
    const int c = i & 511;
    const int t = (i >> 9) & (TT - 1);
    const int b = i >> 18;
    const float* xc = x + (size_t)b * TT * DD;
    const float* wc = w + (size_t)c * KC;
    float acc = bias[c];
    #pragma unroll
    for (int kk = 0; kk < KC; kk++) {
        const int tt = t + kk - (KC - 1) / 2;
        if (tt >= 0 && tt < TT) acc += xc[(size_t)tt * DD + c] * wc[kk];
    }
    y[i] = acc;
}

// batchnorm stats per channel; stats[c]=mean, stats[512+c]=var
__launch_bounds__(256)
__global__ void bn_stats_kernel(const float* __restrict__ x, float* __restrict__ stats)
{
    const int c = blockIdx.x;
    float s = 0.f, ss = 0.f;
    for (int r = threadIdx.x; r < NTOK; r += 256) {
        const float v = x[(size_t)r * DD + c];
        s += v; ss += v * v;
    }
    __shared__ float rs[8], rss[8];
    #pragma unroll
    for (int off = 16; off > 0; off >>= 1) {
        s  += __shfl_xor(s,  off, 32);
        ss += __shfl_xor(ss, off, 32);
    }
    if ((threadIdx.x & 31) == 0) { rs[threadIdx.x >> 5] = s; rss[threadIdx.x >> 5] = ss; }
    __syncthreads();
    if (threadIdx.x == 0) {
        float S = 0.f, SS = 0.f;
        #pragma unroll
        for (int w = 0; w < 8; w++) { S += rs[w]; SS += rss[w]; }
        const float m = S * (1.0f / NTOK);
        stats[c]      = m;
        stats[DD + c] = SS * (1.0f / NTOK) - m * m;
    }
}

// bn apply + swish (in place ok)
__global__ void bn_swish_kernel(const float* __restrict__ x,
                                const float* __restrict__ stats,
                                const float* __restrict__ g,
                                const float* __restrict__ b,
                                float* __restrict__ y)
{
    const int i = blockIdx.x * blockDim.x + threadIdx.x;
    const int c = i & 511;
    const float m   = stats[c];
    const float var = stats[DD + c];
    const float h = (x[i] - m) * rsqrtf(var + LNEPS) * g[c] + b[c];
    y[i] = h * (1.0f / (1.0f + __expf(-h)));
}

// ---------------------------------------------------------------------------
extern "C" void kernel_launch(void* const* d_in, const int* in_sizes, int n_in,
                              void* d_out, int out_size, void* d_ws, size_t ws_size,
                              hipStream_t stream)
{
    (void)in_sizes; (void)n_in; (void)out_size; (void)ws_size;

    const float* inp      = (const float*)d_in[0];
    const float* lin_in_w = (const float*)d_in[1];
    const float* lin_in_b = (const float*)d_in[2];
    const float* ln_in_g  = (const float*)d_in[3];
    const float* ln_in_b  = (const float*)d_in[4];
    const float* ff1_w1   = (const float*)d_in[5];
    const float* ff1_b1   = (const float*)d_in[6];
    const float* ff1_w2   = (const float*)d_in[7];
    const float* ff1_b2   = (const float*)d_in[8];
    const float* ff2_w1   = (const float*)d_in[9];
    const float* ff2_b1   = (const float*)d_in[10];
    const float* ff2_w2   = (const float*)d_in[11];
    const float* ff2_b2   = (const float*)d_in[12];
    const float* wq = (const float*)d_in[13]; const float* bq = (const float*)d_in[14];
    const float* wk = (const float*)d_in[15]; const float* bk = (const float*)d_in[16];
    const float* wv = (const float*)d_in[17]; const float* bv = (const float*)d_in[18];
    const float* wo = (const float*)d_in[19]; const float* bo = (const float*)d_in[20];
    const float* wpos = (const float*)d_in[21];
    const float* pbu  = (const float*)d_in[22];
    const float* pbv  = (const float*)d_in[23];
    const float* pw1_w = (const float*)d_in[24]; const float* pw1_b = (const float*)d_in[25];
    const float* dw_w  = (const float*)d_in[26]; const float* dw_b  = (const float*)d_in[27];
    const float* bn_g  = (const float*)d_in[28]; const float* bn_b  = (const float*)d_in[29];
    const float* pw2_w = (const float*)d_in[30]; const float* pw2_b = (const float*)d_in[31];
    const float* ln_ff1_g  = (const float*)d_in[32]; const float* ln_ff1_b  = (const float*)d_in[33];
    const float* ln_attn_g = (const float*)d_in[34]; const float* ln_attn_b = (const float*)d_in[35];
    const float* ln_ff2_g  = (const float*)d_in[36]; const float* ln_ff2_b  = (const float*)d_in[37];

    float* x = (float*)d_out;                 // (B,T,D) running activation

    // workspace carve-up (floats)
    float* w = (float*)d_ws;
    float* nx    = w; w += (size_t)NTOK * DD;
    float* h1    = w; w += (size_t)NTOK * DFFN;      // also pw1 out
    float* qb    = w; w += (size_t)NTOK * DD;
    float* kb    = w; w += (size_t)NTOK * DD;
    float* vb    = w; w += (size_t)NTOK * DD;
    float* qu    = w; w += (size_t)NTOK * DD;
    float* qv    = w; w += (size_t)NTOK * DD;
    float* attno = w; w += (size_t)NTOK * DD;
    float* glu   = w; w += (size_t)NTOK * DD;
    float* dwb   = w; w += (size_t)NTOK * DD;
    float* p     = w; w += (size_t)TT * DD;
    float* pe    = w; w += (size_t)TT * DD;
    float* stats = w; w += 2 * DD;
    float* ac    = w; w += (size_t)BS * HH * TT * TT;
    float* bd    = w; w += (size_t)BS * HH * TT * TT;

    const int ELT = (NTOK * DD) / 256;
    const int LNG = NTOK / 8;
    const float invsqrt_dk = 0.125f;   // 1/sqrt(64)

    pe_kernel<<<(TT * DD) / 256, 256, 0, stream>>>(pe);

    // input linear + input LN
    launch_gemm(stream, inp, lin_in_w, lin_in_b, nullptr, nx,
                NTOK, DD, DD, DD, DD, DD, 0,0,0,0,0,0, 1, 1, 0, 0, 1.0f, 0);
    ln_kernel<<<LNG, 256, 0, stream>>>(nx, ln_in_g, ln_in_b, x);

    for (int i = 0; i < NL; i++) {
        const float* ff1w1 = ff1_w1 + (size_t)i * DD * DFFN;
        const float* ff1b1 = ff1_b1 + (size_t)i * DFFN;
        const float* ff1w2 = ff1_w2 + (size_t)i * DFFN * DD;
        const float* ff1b2 = ff1_b2 + (size_t)i * DD;
        const float* ff2w1 = ff2_w1 + (size_t)i * DD * DFFN;
        const float* ff2b1 = ff2_b1 + (size_t)i * DFFN;
        const float* ff2w2 = ff2_w2 + (size_t)i * DFFN * DD;
        const float* ff2b2 = ff2_b2 + (size_t)i * DD;
        const float* wqi = wq + (size_t)i * DD * DD; const float* bqi = bq + (size_t)i * DD;
        const float* wki = wk + (size_t)i * DD * DD; const float* bki = bk + (size_t)i * DD;
        const float* wvi = wv + (size_t)i * DD * DD; const float* bvi = bv + (size_t)i * DD;
        const float* woi = wo + (size_t)i * DD * DD; const float* boi = bo + (size_t)i * DD;
        const float* wposi = wpos + (size_t)i * DD * DD;
        const float* pbui = pbu + (size_t)i * DD;
        const float* pbvi = pbv + (size_t)i * DD;
        const float* pw1wi = pw1_w + (size_t)i * 2 * DD * DD;
        const float* pw1bi = pw1_b + (size_t)i * 2 * DD;
        const float* dwwi  = dw_w + (size_t)i * DD * KC;
        const float* dwbi  = dw_b + (size_t)i * DD;
        const float* bngi  = bn_g + (size_t)i * DD;
        const float* bnbi  = bn_b + (size_t)i * DD;
        const float* pw2wi = pw2_w + (size_t)i * DD * DD;
        const float* pw2bi = pw2_b + (size_t)i * DD;
        const float* lf1g = ln_ff1_g + (size_t)i * DD,  *lf1b = ln_ff1_b + (size_t)i * DD;
        const float* latg = ln_attn_g + (size_t)i * DD, *latb = ln_attn_b + (size_t)i * DD;
        const float* lf2g = ln_ff2_g + (size_t)i * DD,  *lf2b = ln_ff2_b + (size_t)i * DD;

        // ---- macaron FFN 1: x += 0.5 * ffn(LN(x))
        ln_kernel<<<LNG, 256, 0, stream>>>(x, lf1g, lf1b, nx);
        launch_gemm(stream, nx, ff1w1, ff1b1, nullptr, h1,
                    NTOK, DFFN, DD, DD, DFFN, DFFN, 0,0,0,0,0,0, 1,1, 0, 1, 1.0f, 0);
        launch_gemm(stream, h1, ff1w2, ff1b2, x, x,
                    NTOK, DD, DFFN, DFFN, DD, DD, 0,0,0,0,0,0, 1,1, 0, 0, 0.5f, 0);

        // ---- rel-pos MHSA: x += attn(LN(x))
        ln_kernel<<<LNG, 256, 0, stream>>>(x, latg, latb, nx);
        launch_gemm(stream, nx, wqi, bqi, nullptr, qb,
                    NTOK, DD, DD, DD, DD, DD, 0,0,0,0,0,0, 1,1, 0, 0, 1.0f, 0);
        launch_gemm(stream, nx, wki, bki, nullptr, kb,
                    NTOK, DD, DD, DD, DD, DD, 0,0,0,0,0,0, 1,1, 0, 0, 1.0f, 0);
        launch_gemm(stream, nx, wvi, bvi, nullptr, vb,
                    NTOK, DD, DD, DD, DD, DD, 0,0,0,0,0,0, 1,1, 0, 0, 1.0f, 0);
        launch_gemm(stream, pe, wposi, nullptr, nullptr, p,
                    TT, DD, DD, DD, DD, DD, 0,0,0,0,0,0, 1,1, 0, 0, 1.0f, 0);
        addpb_kernel<<<ELT, 256, 0, stream>>>(qb, pbui, pbvi, qu, qv);

        // AC[b,h] = (q+pbu) @ K^T      (batched NT, non-temporal C)
        launch_gemm(stream, qu, kb, nullptr, nullptr, ac,
                    TT, TT, DKK, DD, DD, TT,
                    (long)TT * DD, DKK, (long)TT * DD, DKK,
                    (long)HH * TT * TT, (long)TT * TT,
                    BS, HH, 1, 0, 1.0f, 1);
        // BD[b,h] = (q+pbv) @ P_h^T    (batched NT, B shared across b)
        launch_gemm(stream, qv, p, nullptr, nullptr, bd,
                    TT, TT, DKK, DD, DD, TT,
                    (long)TT * DD, DKK, 0L, DKK,
                    (long)HH * TT * TT, (long)TT * TT,
                    BS, HH, 1, 0, 1.0f, 1);
        // softmax((AC + rel_shift(BD)) / sqrt(dk)) in place over ac
        attn_softmax_kernel<<<dim3(TT, BS * HH), 256, 0, stream>>>(ac, bd, ac, invsqrt_dk);
        // attno[b,q,h,:] = attn @ V    (batched NN, N = DK = 64)
        launch_gemm(stream, ac, vb, nullptr, nullptr, attno,
                    TT, DKK, TT, TT, DD, DD,
                    (long)HH * TT * TT, (long)TT * TT,
                    (long)TT * DD, DKK, (long)TT * DD, DKK,
                    BS, HH, 0, 0, 1.0f, 0);
        // output projection + residual
        launch_gemm(stream, attno, woi, boi, x, x,
                    NTOK, DD, DD, DD, DD, DD, 0,0,0,0,0,0, 1,1, 0, 0, 1.0f, 0);

        // ---- conv module (reuses ln_attn per reference): x += conv(LN(x))
        ln_kernel<<<LNG, 256, 0, stream>>>(x, latg, latb, nx);
        launch_gemm(stream, nx, pw1wi, pw1bi, nullptr, h1,          // @ pw1_w.T
                    NTOK, 2 * DD, DD, DD, DD, 2 * DD, 0,0,0,0,0,0, 1,1, 1, 0, 1.0f, 0);
        glu_kernel<<<ELT, 256, 0, stream>>>(h1, glu);
        dwconv_kernel<<<ELT, 256, 0, stream>>>(glu, dwwi, dwbi, dwb);
        bn_stats_kernel<<<DD, 256, 0, stream>>>(dwb, stats);
        bn_swish_kernel<<<ELT, 256, 0, stream>>>(dwb, stats, bngi, bnbi, dwb);
        launch_gemm(stream, dwb, pw2wi, pw2bi, x, x,                // @ pw2_w.T
                    NTOK, DD, DD, DD, DD, DD, 0,0,0,0,0,0, 1,1, 1, 0, 1.0f, 0);

        // ---- FFN 2: x += 0.5 * ffn(LN(x))
        ln_kernel<<<LNG, 256, 0, stream>>>(x, lf2g, lf2b, nx);
        launch_gemm(stream, nx, ff2w1, ff2b1, nullptr, h1,
                    NTOK, DFFN, DD, DD, DFFN, DFFN, 0,0,0,0,0,0, 1,1, 0, 1, 1.0f, 0);
        launch_gemm(stream, h1, ff2w2, ff2b2, x, x,
                    NTOK, DD, DFFN, DFFN, DD, DD, 0,0,0,0,0,0, 1,1, 0, 0, 0.5f, 0);
    }
}
